// Transformer_16209206575272
// MI455X (gfx1250) — matrix operand order repaired and checked by reading, not compiled, not run on hardware
//
#include <hip/hip_runtime.h>
#include <hip/hip_bf16.h>
#include <math.h>

// ---------------------------------------------------------------------------
// MI455X / gfx1250 implementation: all matmuls routed through
// v_wmma_f32_16x16x32_bf16 (fp32 activations truncated to bf16, fp32 accum).
// LDS tiles are stored so every WMMA fragment is two contiguous 16-byte
// ds_load_b128's; staging uses global_load_b128 + packed bf16 converts.
// Shape specialization is compile-time: EXACT kernels carry zero guards.
// ---------------------------------------------------------------------------

typedef __bf16 bf16_t;
typedef __attribute__((ext_vector_type(16))) __bf16 v16bf;
typedef __attribute__((ext_vector_type(8)))  __bf16 v8bf;
typedef __attribute__((ext_vector_type(8)))  float  v8f;

// model dims
constexpr int DM  = 256;
constexpr int NH  = 8;
constexpr int DHD = 32;
constexpr int FFD = 2048;
constexpr int NLAYER = 6;
constexpr int LQ  = 300;
constexpr int BB  = 16;
constexpr int SSL = 1024;
constexpr int RQ  = LQ * BB;    // 4800 query rows
constexpr int RS  = SSL * BB;   // 16384 memory rows
constexpr float ATT_SCALE = 0.17677669529663687f;  // 1/sqrt(32)

#define BM 64
#define BN 64
#define BK 32
#define LPAD 8   // LDS row pad (bf16 elems): 80B pitch -> conflict-free b128 reads

static __device__ __forceinline__ bf16_t f2bf(float f) { return (bf16_t)f; }

static __device__ __forceinline__ v16bf concat8(v8bf a, v8bf b) {
  return __builtin_shufflevector(a, b, 0,1,2,3,4,5,6,7,8,9,10,11,12,13,14,15);
}
static __device__ __forceinline__ v8bf pack8(float4 a, float4 b) {
  v8bf r;
  r[0]=f2bf(a.x); r[1]=f2bf(a.y); r[2]=f2bf(a.z); r[3]=f2bf(a.w);
  r[4]=f2bf(b.x); r[5]=f2bf(b.y); r[6]=f2bf(b.z); r[7]=f2bf(b.w);
  return r;
}
static __device__ __forceinline__ v16bf pack16(float4 a, float4 b, float4 c, float4 d) {
  return concat8(pack8(a, b), pack8(c, d));
}

// ---------------------------------------------------------------------------
// Generic GEMM: C[M,N] = act(A[M,K] @ W[K,N] + bias[N]) (+ addsrc[M,N])
// act: 0 = none, 1 = relu.  8 waves, 64x64 tile, K-step 32, bf16 WMMA.
// EXACT=true: M%64==0, N%64==0, K%32==0, no guards anywhere in the K-loop.
// As row-major [m][k]; W tile stored transposed Bt[n][k] so B fragments are
// contiguous.  80B LDS pitch => bank-conflict-free b128 fragment loads.
// ---------------------------------------------------------------------------
template <bool EXACT>
__global__ __launch_bounds__(256) void gemm_bias_act(
    const float* __restrict__ A, int lda,
    const float* __restrict__ W, int ldw,
    const float* __restrict__ bias,
    const float* __restrict__ addsrc,
    float* __restrict__ C, int ldc,
    int M, int N, int K, int act)
{
  __shared__ bf16_t As[BM][BK + LPAD];
  __shared__ bf16_t Bt[BN][BK + LPAD];

  const int tid   = threadIdx.x;
  const int lane  = tid & 31;
  const int wid   = tid >> 5;
  const int wm    = (wid & 3) * 16;   // wave row offset within tile
  const int wn    = (wid >> 2) * 32;  // wave col offset within tile
  const int m0    = blockIdx.y * BM;
  const int n0    = blockIdx.x * BN;
  const int khalf = lane >> 4;
  const int l16   = lane & 15;

  // staging coords
  const int amr = tid >> 2;            // 0..63 (A row)
  const int akc = (tid & 3) * 8;       // 0/8/16/24 (A col)
  const int bkr = tid >> 3;            // 0..31 (W row = k)
  const int bnc = (tid & 7) * 8;       // 0..56 (W col = n)

  const float* aptr = A + (size_t)(m0 + amr) * lda + akc;
  const float* wptr = W + (size_t)bkr * ldw + n0 + bnc;
  const size_t wstep = (size_t)BK * ldw;

  v8f c0 = {}; v8f c1 = {};

  for (int k0 = 0; k0 < K; k0 += BK) {
    if constexpr (EXACT) {
      // ---- guard-free staging: two b128 loads each for A and W ----
      const float4* ap4 = (const float4*)(aptr + k0);
      *(v8bf*)&As[amr][akc] = pack8(ap4[0], ap4[1]);

      const float* bp = wptr + (size_t)k0 * ldw;
      const float4* bp4 = (const float4*)bp;
      float4 x0 = bp4[0], x1 = bp4[1];
      Bt[bnc + 0][bkr] = f2bf(x0.x); Bt[bnc + 1][bkr] = f2bf(x0.y);
      Bt[bnc + 2][bkr] = f2bf(x0.z); Bt[bnc + 3][bkr] = f2bf(x0.w);
      Bt[bnc + 4][bkr] = f2bf(x1.x); Bt[bnc + 5][bkr] = f2bf(x1.y);
      Bt[bnc + 6][bkr] = f2bf(x1.z); Bt[bnc + 7][bkr] = f2bf(x1.w);

      // prefetch next K tile into cache (global_prefetch_b8)
      if (k0 + BK < K) {
        __builtin_prefetch(aptr + k0 + BK, 0, 1);
        __builtin_prefetch(bp + wstep, 0, 1);
      }
    } else {
      // ---- guarded staging (small conf-MLP GEMMs only) ----
      {
        const float* ap = aptr + k0;
        const bool mok = (m0 + amr) < M;
#pragma unroll
        for (int j = 0; j < 8; ++j) {
          float v = (mok && (k0 + akc + j) < K) ? ap[j] : 0.0f;
          As[amr][akc + j] = f2bf(v);
        }
      }
      {
        const float* bp = wptr + (size_t)k0 * ldw;
        const bool kok = (k0 + bkr) < K;
#pragma unroll
        for (int j = 0; j < 8; ++j) {
          float v = (kok && (n0 + bnc + j) < N) ? bp[j] : 0.0f;
          Bt[bnc + j][bkr] = f2bf(v);
        }
      }
    }
    __syncthreads();

    // A fragment: rows wm+l16; k spans [khalf*8, +8) and [16+khalf*8, +8)
    const v8bf* arow = (const v8bf*)&As[wm + l16][0];
    v16bf af = concat8(arow[khalf], arow[2 + khalf]);
    // B fragments: cols wn+l16 / wn+16+l16; k spans [khalf*8, +8) and [16+khalf*8, +8)
    const v8bf* br0 = (const v8bf*)&Bt[wn + l16][0];
    const v8bf* br1 = (const v8bf*)&Bt[wn + 16 + l16][0];
    v16bf bf0 = concat8(br0[khalf], br0[2 + khalf]);
    v16bf bf1 = concat8(br1[khalf], br1[2 + khalf]);

    c0 = __builtin_amdgcn_wmma_f32_16x16x32_bf16(false, af, false, bf0, (short)0, c0, false, false);
    c1 = __builtin_amdgcn_wmma_f32_16x16x32_bf16(false, af, false, bf1, (short)0, c1, false, false);
    __syncthreads();
  }

  // epilogue: C layout = lane col, VGPR row (row m = r + 8*khalf)
#pragma unroll
  for (int t = 0; t < 2; ++t) {
    v8f cc = t ? c1 : c0;
    int col = n0 + wn + t * 16 + l16;
    if (EXACT || col < N) {
      float bv = bias ? bias[col] : 0.0f;
#pragma unroll
      for (int r = 0; r < 8; ++r) {
        int m = m0 + wm + r + 8 * khalf;
        if (EXACT || m < M) {
          float v = cc[r] + bv;
          if (act == 1) v = fmaxf(v, 0.0f);
          if (addsrc) v += addsrc[(size_t)m * ldc + col];
          C[(size_t)m * ldc + col] = v;
        }
      }
    }
  }
}

// ---------------------------------------------------------------------------
// Gated flash attention: one wave per (16-query block, b, h).
// Qp/Kp/Vp are pre-gated projections [L,B,H,DH] fp32; Gv = sigmoid(vc) [S,B,H].
// O[Lq,B,H,DH] normalized output, Ni[Lq,B,H] = softmax(A) @ gv.
// DH = 32 -> one WMMA per 16x16 score tile; 32-key steps.
// ---------------------------------------------------------------------------
__global__ __launch_bounds__(32) void attn_gated_kernel(
    const float* __restrict__ Qp, const float* __restrict__ Kp,
    const float* __restrict__ Vp, const float* __restrict__ Gv,
    float* __restrict__ O, float* __restrict__ Ni,
    int Lq_, int Slen, float scale)
{
  __shared__ float  Sbuf[16][33];
  __shared__ bf16_t Pbuf[16][BK + LPAD];   // [query][key]
  __shared__ bf16_t Vt[DHD][BK + LPAD];    // transposed V tile [dh][key]
  __shared__ float  gvs[32];
  __shared__ float  mbuf[16], lbuf[16], cbuf[16], nbuf[16];

  const int lane  = threadIdx.x;
  const int l16   = lane & 15;
  const int khalf = lane >> 4;
  const int b     = blockIdx.y / NH;
  const int h     = blockIdx.y % NH;
  const int q0    = blockIdx.x * 16;

  if (lane < 16) { mbuf[lane] = -1e30f; lbuf[lane] = 0.0f; nbuf[lane] = 0.0f; }

  // Q fragment (A layout): row l16, k spans [khalf*8,+8) and [16+khalf*8,+8)
  int ql = q0 + l16; if (ql >= Lq_) ql = Lq_ - 1;
  const float4* q4 = (const float4*)(Qp + ((size_t)(ql * BB + b) * NH + h) * DHD);
  v16bf qa = pack16(q4[2 * khalf], q4[2 * khalf + 1],
                    q4[4 + 2 * khalf], q4[5 + 2 * khalf]);
  v8f o0 = {}; v8f o1 = {};
  __syncthreads();

  for (int s0 = 0; s0 < Slen; s0 += 32) {
    // ---- stage gv + transposed V tile (each lane: one key row, contiguous) ----
    {
      int sg = s0 + lane;
      gvs[lane] = (sg < Slen) ? Gv[(size_t)(sg * BB + b) * NH + h] : 0.0f;
      int sc = (sg < Slen) ? sg : (Slen - 1);
      const float4* vr = (const float4*)(Vp + ((size_t)(sc * BB + b) * NH + h) * DHD);
#pragma unroll
      for (int d4 = 0; d4 < 8; ++d4) {
        float4 x = vr[d4];
        Vt[d4 * 4 + 0][lane] = f2bf(x.x);
        Vt[d4 * 4 + 1][lane] = f2bf(x.y);
        Vt[d4 * 4 + 2][lane] = f2bf(x.z);
        Vt[d4 * 4 + 3][lane] = f2bf(x.w);
      }
    }
    // ---- K^T fragments (B layout): lane = key column, dh spans [khalf*8,+8) and [16+khalf*8,+8) ----
    v16bf kb0, kb1;
    {
      int sA = s0 + l16;      if (sA >= Slen) sA = Slen - 1;
      int sB = s0 + 16 + l16; if (sB >= Slen) sB = Slen - 1;
      const float4* kp0 = (const float4*)(Kp + ((size_t)(sA * BB + b) * NH + h) * DHD + khalf * 8);
      const float4* kp1 = (const float4*)(Kp + ((size_t)(sB * BB + b) * NH + h) * DHD + khalf * 8);
      kb0 = pack16(kp0[0], kp0[1], kp0[4], kp0[5]);
      kb1 = pack16(kp1[0], kp1[1], kp1[4], kp1[5]);
    }
    v8f zz = {};
    v8f sc0 = __builtin_amdgcn_wmma_f32_16x16x32_bf16(false, qa, false, kb0, (short)0, zz, false, false);
    v8f sc1 = __builtin_amdgcn_wmma_f32_16x16x32_bf16(false, qa, false, kb1, (short)0, zz, false, false);

    // spill scores to LDS with key masking
    {
      bool v0 = (s0 + l16) < Slen;
      bool v1 = (s0 + 16 + l16) < Slen;
#pragma unroll
      for (int r = 0; r < 8; ++r) {
        int m = r + 8 * khalf;
        Sbuf[m][l16]      = v0 ? sc0[r] * scale : -1e30f;
        Sbuf[m][16 + l16] = v1 ? sc1[r] * scale : -1e30f;
      }
    }
    __syncthreads();

    // online softmax rows (lanes 0..15, one row each)
    if (lane < 16) {
      const int m = lane;
      float mOld = mbuf[m], mNew = mOld;
#pragma unroll
      for (int j = 0; j < 32; ++j) mNew = fmaxf(mNew, Sbuf[m][j]);
      float corr = __expf(mOld - mNew);
      float ps = 0.0f, ns = 0.0f;
#pragma unroll
      for (int j = 0; j < 32; ++j) {
        float p = __expf(Sbuf[m][j] - mNew);
        Pbuf[m][j] = f2bf(p);
        ps += p; ns += p * gvs[j];
      }
      mbuf[m] = mNew;
      lbuf[m] = lbuf[m] * corr + ps;
      nbuf[m] = nbuf[m] * corr + ns;
      cbuf[m] = corr;
    }
    __syncthreads();

    // rescale running O
#pragma unroll
    for (int r = 0; r < 8; ++r) {
      float corr = cbuf[r + 8 * khalf];
      o0[r] *= corr; o1[r] *= corr;
    }
    // P fragment (A layout, K = 32 keys): two contiguous b128 spans
    const v8bf* pr = (const v8bf*)&Pbuf[l16][0];
    v16bf pa = concat8(pr[khalf], pr[2 + khalf]);
    // V fragments (B layout) from transposed tile: contiguous over keys
    const v8bf* vt0 = (const v8bf*)&Vt[l16][0];
    const v8bf* vt1 = (const v8bf*)&Vt[16 + l16][0];
    v16bf vb0 = concat8(vt0[khalf], vt0[2 + khalf]);
    v16bf vb1 = concat8(vt1[khalf], vt1[2 + khalf]);

    o0 = __builtin_amdgcn_wmma_f32_16x16x32_bf16(false, pa, false, vb0, (short)0, o0, false, false);
    o1 = __builtin_amdgcn_wmma_f32_16x16x32_bf16(false, pa, false, vb1, (short)0, o1, false, false);
    __syncthreads();   // protects Pbuf/Vt/Sbuf for next iteration
  }

  // normalize + store
#pragma unroll
  for (int r = 0; r < 8; ++r) {
    int m = r + 8 * khalf;
    int l = q0 + m;
    if (l < Lq_) {
      float inv = 1.0f / lbuf[m];
      float* op = O + ((size_t)(l * BB + b) * NH + h) * DHD;
      op[l16]      = o0[r] * inv;
      op[16 + l16] = o1[r] * inv;
    }
  }
  if (lane < 16) {
    int l = q0 + lane;
    if (l < Lq_) Ni[(size_t)(l * BB + b) * NH + h] = nbuf[lane] / lbuf[lane];
  }
}

// ---------------------------------------------------------------------------
// LayerNorm over D=256, one row per 256-thread block; optional fused residual.
// ---------------------------------------------------------------------------
__global__ __launch_bounds__(256) void ln_kernel(
    const float* __restrict__ x, const float* __restrict__ res,
    const float* __restrict__ g, const float* __restrict__ bta,
    float* __restrict__ out)
{
  __shared__ float sbuf[256];
  const int row = blockIdx.x;
  const int t   = threadIdx.x;
  float v = x[(size_t)row * DM + t];
  if (res) v += res[(size_t)row * DM + t];

  sbuf[t] = v; __syncthreads();
  for (int off = 128; off > 0; off >>= 1) { if (t < off) sbuf[t] += sbuf[t + off]; __syncthreads(); }
  float mean = sbuf[0] * (1.0f / DM);
  __syncthreads();
  float d = v - mean;
  sbuf[t] = d * d; __syncthreads();
  for (int off = 128; off > 0; off >>= 1) { if (t < off) sbuf[t] += sbuf[t + off]; __syncthreads(); }
  float var = sbuf[0] * (1.0f / DM);
  __syncthreads();
  out[(size_t)row * DM + t] = d * rsqrtf(var + 1e-5f) * g[t] + bta[t];
}

// ---------------------------------------------------------------------------
// elementwise helpers
// ---------------------------------------------------------------------------
__global__ void add_kernel(const float* __restrict__ a, const float* __restrict__ b,
                           float* __restrict__ c, int n)
{
  int i = blockIdx.x * 256 + threadIdx.x;
  if (i < n) c[i] = a[i] + b[i];
}

__global__ void sigmoid_kernel(const float* __restrict__ x, float* __restrict__ y, int n)
{
  int i = blockIdx.x * 256 + threadIdx.x;
  if (i < n) y[i] = 1.0f / (1.0f + __expf(-x[i]));
}

// x[rows, H, DH] *= sigmoid(conf[rows, H]); D=256, DH=32 hard-coded
__global__ void gate_kernel(float* __restrict__ x, const float* __restrict__ conf, int n)
{
  int i = blockIdx.x * 256 + threadIdx.x;
  if (i < n) {
    int r  = i >> 8;
    int hh = (i & 255) >> 5;
    x[i] *= 1.0f / (1.0f + __expf(-conf[r * NH + hh]));
  }
}

// ---------------------------------------------------------------------------
// input index map: dict order = 6 tensors, then params sorted by name
// ---------------------------------------------------------------------------
enum {
  IN_TGT = 0, IN_MEMORY, IN_POS, IN_QPOS, IN_INTER, IN_IMG_INTER,
  P_CA_WO, P_CA_WQKV, P_CA_BO, P_CA_BQKV,
  P_FNORM_B, P_FNORM_G,
  P_KCONF1_W1, P_KCONF1_W2, P_KCONF1_B1, P_KCONF1_B2,
  P_KCONF2_W1, P_KCONF2_W2, P_KCONF2_B1, P_KCONF2_B2,
  P_KPCONF1_W1, P_KPCONF1_W2, P_KPCONF1_B1, P_KPCONF1_B2,
  P_KPCONF2_W1, P_KPCONF2_W2, P_KPCONF2_B1, P_KPCONF2_B2,
  P_LIN1_W, P_LIN1_B, P_LIN2_W, P_LIN2_B,
  P_NORM1_B, P_NORM1_G, P_NORM2_B, P_NORM2_G, P_NORM3_B, P_NORM3_G,
  P_QCONF1_W1, P_QCONF1_W2, P_QCONF1_B1, P_QCONF1_B2,
  P_QCONF2_W1, P_QCONF2_W2, P_QCONF2_B1, P_QCONF2_B2,
  P_QPCONF1_W1, P_QPCONF1_W2, P_QPCONF1_B1, P_QPCONF1_B2,
  P_QPCONF2_W1, P_QPCONF2_W2, P_QPCONF2_B1, P_QPCONF2_B2,
  P_SA_WO, P_SA_WQKV, P_SA_BO, P_SA_BQKV,
  P_UPD0_W1, P_UPD0_W2, P_UPD0_B1, P_UPD0_B2,
  P_UPD1_W1, P_UPD1_W2, P_UPD1_B1, P_UPD1_B2,
  P_UPD2_W1, P_UPD2_W2, P_UPD2_B1, P_UPD2_B2,
  P_VCONF1_W1, P_VCONF1_W2, P_VCONF1_B1, P_VCONF1_B2,
  P_VCONF2_W1, P_VCONF2_W2, P_VCONF2_B1, P_VCONF2_B2
};

extern "C" void kernel_launch(void* const* d_in, const int* in_sizes, int n_in,
                              void* d_out, int out_size, void* d_ws, size_t ws_size,
                              hipStream_t stream)
{
  (void)in_sizes; (void)n_in; (void)out_size; (void)ws_size;

  const float* tgt      = (const float*)d_in[IN_TGT];
  const float* memory   = (const float*)d_in[IN_MEMORY];
  const float* pos      = (const float*)d_in[IN_POS];
  const float* qpos     = (const float*)d_in[IN_QPOS];
  const float* inter_in = (const float*)d_in[IN_INTER];
  const float* img_int  = (const float*)d_in[IN_IMG_INTER];
  float* outp = (float*)d_out;

  // per-layer slice helper
  auto L = [&](int pidx, int i, size_t per) -> const float* {
    return (const float*)d_in[pidx] + (size_t)i * per;
  };

  // workspace carve-up (fp32)
  float* w = (float*)d_ws;
  size_t off = 0;
  auto alloc = [&](size_t n) { float* p = w + off; off += n; return p; };
  float* cur     = alloc((size_t)RQ * DM);
  float* qkb     = alloc((size_t)RQ * DM);
  float* mem_pos = alloc((size_t)RS * DM);
  float* qp      = alloc((size_t)RS * DM);
  float* kp      = alloc((size_t)RS * DM);
  float* vp      = alloc((size_t)RS * DM);
  float* hid     = alloc((size_t)RS * DM);
  float* ffnh    = alloc((size_t)RQ * FFD);
  float* attn_o  = alloc((size_t)RQ * DM);
  float* t2      = alloc((size_t)RQ * DM);
  float* pconf   = alloc((size_t)RS);
  float* confin  = alloc((size_t)RS);
  float* qcb     = alloc((size_t)RS * NH);
  float* kcb     = alloc((size_t)RS * NH);
  float* vcb     = alloc((size_t)RS * NH);
  float* gvb     = alloc((size_t)RS * NH);
  float* nib     = alloc((size_t)RQ * NH);
  float* int0    = alloc((size_t)RQ);
  float* int1    = alloc((size_t)RQ);

  auto gemm = [&](const float* A, int lda, const float* W, int ldw, const float* bias,
                  const float* addsrc, float* C, int ldc, int M, int N, int K, int act) {
    dim3 g((N + BN - 1) / BN, (M + BM - 1) / BM);
    const bool exact = (M % BM == 0) && (N % BN == 0) && (K % BK == 0);
    if (exact)
      gemm_bias_act<true><<<g, 256, 0, stream>>>(A, lda, W, ldw, bias, addsrc, C, ldc, M, N, K, act);
    else
      gemm_bias_act<false><<<g, 256, 0, stream>>>(A, lda, W, ldw, bias, addsrc, C, ldc, M, N, K, act);
  };
  // y = relu(x@W1 + b1)@W2 + b2 (+addsrc); W1:[Kin,256], W2:[256,Nout]
  auto mlp2 = [&](const float* x, int M, int Kin, const float* W1, const float* b1,
                  const float* W2, const float* b2, int Nout, float* y, const float* addsrc) {
    gemm(x, Kin, W1, DM, b1, nullptr, hid, DM, M, DM, Kin, 1);
    gemm(hid, DM, W2, Nout, b2, addsrc, y, Nout, M, Nout, DM, 0);
  };
  auto addk = [&](const float* a, const float* b, float* c, int n) {
    add_kernel<<<(n + 255) / 256, 256, 0, stream>>>(a, b, c, n);
  };
  auto gate = [&](float* x, const float* conf, int rows) {
    int n = rows * DM;
    gate_kernel<<<(n + 255) / 256, 256, 0, stream>>>(x, conf, n);
  };
  auto lnk = [&](const float* x, const float* res, const float* g, const float* b,
                 float* o, int rows) {
    ln_kernel<<<rows, 256, 0, stream>>>(x, res, g, b, o);
  };

  // init: cur = tgt, inter = interactiveness, mem_pos = memory + pos (loop-invariant)
  hipMemcpyAsync(cur,  tgt,      (size_t)RQ * DM * sizeof(float), hipMemcpyDeviceToDevice, stream);
  hipMemcpyAsync(int0, inter_in, (size_t)RQ * sizeof(float),      hipMemcpyDeviceToDevice, stream);
  addk(memory, pos, mem_pos, RS * DM);

  float* interCur = int0;
  float* interNext = int1;
  const int qblocks = (LQ + 15) / 16;

  for (int i = 0; i < NLAYER; ++i) {
    // ---------------- gated self-attention confs ----------------
    mlp2(qpos, RQ, DM, L(P_QPCONF1_W1, i, 65536), L(P_QPCONF1_B1, i, 256),
         L(P_QPCONF1_W2, i, 256), L(P_QPCONF1_B2, i, 1), 1, pconf, nullptr);
    addk(interCur, pconf, confin, RQ);
    mlp2(confin, RQ, 1, L(P_QCONF1_W1, i, 256), L(P_QCONF1_B1, i, 256),
         L(P_QCONF1_W2, i, 2048), L(P_QCONF1_B2, i, 8), NH, qcb, nullptr);

    mlp2(qpos, RQ, DM, L(P_KPCONF1_W1, i, 65536), L(P_KPCONF1_B1, i, 256),
         L(P_KPCONF1_W2, i, 256), L(P_KPCONF1_B2, i, 1), 1, pconf, nullptr);
    addk(interCur, pconf, confin, RQ);
    mlp2(confin, RQ, 1, L(P_KCONF1_W1, i, 256), L(P_KCONF1_B1, i, 256),
         L(P_KCONF1_W2, i, 2048), L(P_KCONF1_B2, i, 8), NH, kcb, nullptr);

    mlp2(interCur, RQ, 1, L(P_VCONF1_W1, i, 256), L(P_VCONF1_B1, i, 256),
         L(P_VCONF1_W2, i, 2048), L(P_VCONF1_B2, i, 8), NH, vcb, nullptr);

    // ---------------- self-attention ----------------
    addk(cur, qpos, qkb, RQ * DM);
    {
      const float* Wqkv = L(P_SA_WQKV, i, (size_t)DM * 3 * DM);
      const float* bqkv = L(P_SA_BQKV, i, 3 * DM);
      gemm(qkb, DM, Wqkv + 0,      3 * DM, bqkv + 0,      nullptr, qp, DM, RQ, DM, DM, 0);
      gemm(qkb, DM, Wqkv + DM,     3 * DM, bqkv + DM,     nullptr, kp, DM, RQ, DM, DM, 0);
      gemm(cur, DM, Wqkv + 2 * DM, 3 * DM, bqkv + 2 * DM, nullptr, vp, DM, RQ, DM, DM, 0);
    }
    gate(qp, qcb, RQ);
    gate(kp, kcb, RQ);
    gate(vp, vcb, RQ);
    sigmoid_kernel<<<(RQ * NH + 255) / 256, 256, 0, stream>>>(vcb, gvb, RQ * NH);
    attn_gated_kernel<<<dim3(qblocks, BB * NH), 32, 0, stream>>>(
        qp, kp, vp, gvb, attn_o, nib, LQ, LQ, ATT_SCALE);
    gemm(attn_o, DM, L(P_SA_WO, i, 65536), DM, L(P_SA_BO, i, 256), nullptr, t2, DM, RQ, DM, DM, 0);
    lnk(cur, t2, L(P_NORM1_G, i, 256), L(P_NORM1_B, i, 256), cur, RQ);
    mlp2(nib, RQ, NH, L(P_UPD0_W1, i, 2048), L(P_UPD0_B1, i, 256),
         L(P_UPD0_W2, i, 256), L(P_UPD0_B2, i, 1), 1, interNext, interCur);
    { float* t = interCur; interCur = interNext; interNext = t; }

    // ---------------- gated cross-attention confs ----------------
    mlp2(qpos, RQ, DM, L(P_QPCONF2_W1, i, 65536), L(P_QPCONF2_B1, i, 256),
         L(P_QPCONF2_W2, i, 256), L(P_QPCONF2_B2, i, 1), 1, pconf, nullptr);
    addk(interCur, pconf, confin, RQ);
    mlp2(confin, RQ, 1, L(P_QCONF2_W1, i, 256), L(P_QCONF2_B1, i, 256),
         L(P_QCONF2_W2, i, 2048), L(P_QCONF2_B2, i, 8), NH, qcb, nullptr);

    mlp2(pos, RS, DM, L(P_KPCONF2_W1, i, 65536), L(P_KPCONF2_B1, i, 256),
         L(P_KPCONF2_W2, i, 256), L(P_KPCONF2_B2, i, 1), 1, pconf, nullptr);
    addk(img_int, pconf, confin, RS);
    mlp2(confin, RS, 1, L(P_KCONF2_W1, i, 256), L(P_KCONF2_B1, i, 256),
         L(P_KCONF2_W2, i, 2048), L(P_KCONF2_B2, i, 8), NH, kcb, nullptr);

    mlp2(img_int, RS, 1, L(P_VCONF2_W1, i, 256), L(P_VCONF2_B1, i, 256),
         L(P_VCONF2_W2, i, 2048), L(P_VCONF2_B2, i, 8), NH, vcb, nullptr);

    // ---------------- cross-attention ----------------
    addk(cur, qpos, qkb, RQ * DM);
    {
      const float* Wqkv = L(P_CA_WQKV, i, (size_t)DM * 3 * DM);
      const float* bqkv = L(P_CA_BQKV, i, 3 * DM);
      gemm(qkb,     DM, Wqkv + 0,      3 * DM, bqkv + 0,      nullptr, qp, DM, RQ, DM, DM, 0);
      gemm(mem_pos, DM, Wqkv + DM,     3 * DM, bqkv + DM,     nullptr, kp, DM, RS, DM, DM, 0);
      gemm(memory,  DM, Wqkv + 2 * DM, 3 * DM, bqkv + 2 * DM, nullptr, vp, DM, RS, DM, DM, 0);
    }
    gate(qp, qcb, RQ);
    gate(kp, kcb, RS);
    gate(vp, vcb, RS);
    sigmoid_kernel<<<(RS * NH + 255) / 256, 256, 0, stream>>>(vcb, gvb, RS * NH);
    attn_gated_kernel<<<dim3(qblocks, BB * NH), 32, 0, stream>>>(
        qp, kp, vp, gvb, attn_o, nib, LQ, SSL, ATT_SCALE);
    gemm(attn_o, DM, L(P_CA_WO, i, 65536), DM, L(P_CA_BO, i, 256), nullptr, t2, DM, RQ, DM, DM, 0);
    lnk(cur, t2, L(P_NORM2_G, i, 256), L(P_NORM2_B, i, 256), cur, RQ);
    mlp2(nib, RQ, NH, L(P_UPD1_W1, i, 2048), L(P_UPD1_B1, i, 256),
         L(P_UPD1_W2, i, 256), L(P_UPD1_B2, i, 1), 1, interNext, interCur);
    { float* t = interCur; interCur = interNext; interNext = t; }

    // ---------------- FFN ----------------
    gemm(cur,  DM,  L(P_LIN1_W, i, (size_t)DM * FFD), FFD, L(P_LIN1_B, i, FFD), nullptr,
         ffnh, FFD, RQ, FFD, DM, 1);
    gemm(ffnh, FFD, L(P_LIN2_W, i, (size_t)FFD * DM), DM,  L(P_LIN2_B, i, DM),  nullptr,
         t2,   DM,  RQ, DM, FFD, 0);
    lnk(cur, t2, L(P_NORM3_G, i, 256), L(P_NORM3_B, i, 256), cur, RQ);
    mlp2(cur, RQ, DM, L(P_UPD2_W1, i, 65536), L(P_UPD2_B1, i, 256),
         L(P_UPD2_W2, i, 256), L(P_UPD2_B2, i, 1), 1, interNext, interCur);
    { float* t = interCur; interCur = interNext; interNext = t; }

    // ---------------- per-layer outputs ----------------
    lnk(cur, nullptr, (const float*)d_in[P_FNORM_G], (const float*)d_in[P_FNORM_B],
        outp + (size_t)i * RQ * DM, RQ);
    hipMemcpyAsync(outp + (size_t)NLAYER * RQ * DM + (size_t)i * RQ, interCur,
                   (size_t)RQ * sizeof(float), hipMemcpyDeviceToDevice, stream);
  }
}
